// STFF_v2_60919816126478
// MI455X (gfx1250) — matrix operand order repaired and checked by reading, not compiled, not run on hardware
//
#include <hip/hip_runtime.h>

typedef unsigned short u16;
typedef unsigned int   u32;

typedef __attribute__((ext_vector_type(16))) __bf16 v16bf;
typedef __attribute__((ext_vector_type(8)))  float  v8f;

#define H_IN   38
#define W_IN   30
#define HP     40
#define WPAD   32
#define BATCH  4
#define TSTEPS 100
#define HID1   128
#define HID2   32
#define CIN    160               // padded combined input channels (both layers)
#define KDIM   (CIN*9)           // 1440
#define NKB    (KDIM/32)         // 45 K-steps
#define NPOS   (BATCH*H_IN*W_IN) // 4560
#define NS     4608              // padded N stride (72*64)
#define M1     (4*HID1)          // 512
#define M2     (4*HID2)          // 128

#define H1OUT  77
#define W1OUT  61
#define H2OUT  157
#define W2OUT  125
#define HPOOL  135
#define WPOOL  103

__device__ __forceinline__ u16 f2bf(float f) {
    u32 u = __float_as_uint(f);
    u32 r = u + 0x7FFFu + ((u >> 16) & 1u);
    return (u16)(r >> 16);
}
__device__ __forceinline__ float sigm(float v) { return 1.0f / (1.0f + expf(-v)); }

// CDNA5 async memory->LDS copy (ASYNCcnt-tracked), GV addressing mode.
__device__ __forceinline__ void async_b128(u32 lds_off, const void* gaddr) {
    asm volatile("global_load_async_to_lds_b128 %0, %1, off"
                 :: "v"(lds_off), "v"(gaddr) : "memory");
}

// ---------------------------------------------------------------- utilities
__global__ void stff_zero_u16(u16* __restrict__ p, int n) {
    int i = blockIdx.x * 256 + threadIdx.x;
    if (i < n) p[i] = 0;
}
__global__ void stff_zero_f32(float* __restrict__ p, int n) {
    int i = blockIdx.x * 256 + threadIdx.x;
    if (i < n) p[i] = 0.0f;
}

// ------------------------------------------------------- weight repack (bf16)
__global__ void stff_prep_w1(const float* __restrict__ Wx1, const float* __restrict__ Wh1,
                             u16* __restrict__ Wcat) {
    int tid = blockIdx.x * 256 + threadIdx.x;
    if (tid >= M1 * KDIM) return;
    int m = tid / KDIM, k = tid - m * KDIM;
    int c = k / 9, j = k - c * 9;
    float v = 0.0f;
    if (c == 0)        v = Wx1[m * 9 + j];
    else if (c <= 128) v = Wh1[(m * 128 + (c - 1)) * 9 + j];
    Wcat[tid] = f2bf(v);
}
__global__ void stff_prep_w2(const float* __restrict__ Wx2, const float* __restrict__ Wh2,
                             u16* __restrict__ Wcat) {
    int tid = blockIdx.x * 256 + threadIdx.x;
    if (tid >= M2 * KDIM) return;
    int m = tid / KDIM, k = tid - m * KDIM;
    int c = k / 9, j = k - c * 9;
    float v;
    if (c < 128) v = Wx2[(m * 128 + c) * 9 + j];
    else         v = Wh2[(m * 32 + (c - 128)) * 9 + j];
    Wcat[tid] = f2bf(v);
}

// ------------------------------------------------------------ x_t -> I1 chan0
__global__ void stff_stage_x(const float* __restrict__ x, u16* __restrict__ I1, int t) {
    int n = blockIdx.x * 256 + threadIdx.x;
    if (n >= NPOS) return;
    int b = n / (H_IN * W_IN);
    int pos = n - b * (H_IN * W_IN);
    int y = pos / W_IN, xx = pos - y * W_IN;
    float v = x[(b * TSTEPS + t) * (H_IN * W_IN) + pos];
    I1[((b * CIN + 0) * HP + y + 1) * WPAD + xx + 1] = f2bf(v);
}

// ---------------------------------------------------- im2col: Ibuf -> Bcol[n][k]
// Bcol row n (NS rows, KDIM cols): k = c*9 + r*3 + s -> Ibuf[b][c][y+r][x+s]
__global__ void stff_im2col(const u16* __restrict__ Ibuf, u16* __restrict__ Bc) {
    int tid = blockIdx.x * 256 + threadIdx.x;   // n*CIN + c
    if (tid >= NS * CIN) return;
    int n = tid / CIN;
    int c = tid - n * CIN;
    u16 vals[9];
    if (n < NPOS) {
        int b = n / (H_IN * W_IN);
        int rem = n - b * (H_IN * W_IN);
        int y = rem / W_IN, xx = rem - y * W_IN;
        const u16* base = Ibuf + ((b * CIN + c) * HP + y) * WPAD + xx;
#pragma unroll
        for (int r = 0; r < 3; ++r)
#pragma unroll
            for (int s = 0; s < 3; ++s)
                vals[r * 3 + s] = base[r * WPAD + s];
    } else {
#pragma unroll
        for (int j = 0; j < 9; ++j) vals[j] = 0;
    }
    u16* dst = Bc + (size_t)n * KDIM + c * 9;
#pragma unroll
    for (int j = 0; j < 9; ++j) dst[j] = vals[j];
}

// -------------------------------------------------- WMMA streaming GEMM gates
// G[M x NS] = A[M x 1440] * Bcol^T   (both operands [row][k], bf16 -> f32)
union FragBF { v16bf v; u32 u[8]; };

__global__ __launch_bounds__(128)
void stff_gate_gemm(const u16* __restrict__ A, const u16* __restrict__ Bc,
                    float* __restrict__ Gout) {
    __shared__ __align__(16) u16 sA[2][64 * 32];   // [buf][m_local][k_local]
    __shared__ __align__(16) u16 sB[2][64 * 32];   // [buf][n_local][k_local]

    const int tid  = threadIdx.x;
    const int lane = tid & 31;
    const int wave = tid >> 5;
    const int m0 = blockIdx.x * 64;
    const int n0 = blockIdx.y * 64;

    // staging: thread -> (row 0..63, 16-wide k half); 32B of A + 32B of B per step
    const int row  = tid >> 1;
    const int half = tid & 1;
    const u16* gA = A  + (size_t)(m0 + row) * KDIM + half * 16;
    const u16* gB = Bc + (size_t)(n0 + row) * KDIM + half * 16;
    const int soff = row * 32 + half * 16;
    const u32 lA[2] = { (u32)(uintptr_t)&sA[0][soff], (u32)(uintptr_t)&sA[1][soff] };
    const u32 lB[2] = { (u32)(uintptr_t)&sB[0][soff], (u32)(uintptr_t)&sB[1][soff] };

    v8f zero8 = {};
    v8f acc[4] = {zero8, zero8, zero8, zero8};

    // prologue: stage k-block 0 into buffer 0
    {
        async_b128(lA[0],      gA);
        async_b128(lA[0] + 16, gA + 8);
        async_b128(lB[0],      gB);
        async_b128(lB[0] + 16, gB + 8);
    }

    for (int kb = 0; kb < NKB; ++kb) {
        const int cur = kb & 1;
        if (kb + 1 < NKB) {
            // stage next k-block into the other buffer, then wait for current
            const int nxt = cur ^ 1;
            const u16* pa = gA + (kb + 1) * 32;
            const u16* pb = gB + (kb + 1) * 32;
            async_b128(lA[nxt],      pa);
            async_b128(lA[nxt] + 16, pa + 8);
            async_b128(lB[nxt],      pb);
            async_b128(lB[nxt] + 16, pb + 8);
            // 8 outstanding per wave; async loads retire in order, so <=4 left
            // means the current buffer's 4 are complete.
            asm volatile("s_wait_asynccnt 0x4" ::: "memory");
        } else {
            asm volatile("s_wait_asynccnt 0x0" ::: "memory");
        }
        __syncthreads();

        // A fragment (16x32 bf16, ISA lane layout)
        FragBF a;
        {
            const int hi = lane >> 4;
            const u32* ar = (const u32*)(&sA[cur][(16 * wave + (lane & 15)) * 32]);
#pragma unroll
            for (int i = 0; i < 4; ++i) a.u[i] = ar[i + 4 * hi];
#pragma unroll
            for (int i = 4; i < 8; ++i) a.u[i] = ar[(i - 4) + 8 + 4 * hi];
        }
        // B fragments (32x16 bf16) + 4 WMMAs
        {
            const int hi = lane >> 4;
#pragma unroll
            for (int nsub = 0; nsub < 4; ++nsub) {
                FragBF bfr;
                const u32* br = (const u32*)(&sB[cur][(nsub * 16 + (lane & 15)) * 32]);
#pragma unroll
                for (int i = 0; i < 8; ++i) bfr.u[i] = br[(i & 3) + ((i >> 2) << 3) + 4 * hi];
                acc[nsub] = __builtin_amdgcn_wmma_f32_16x16x32_bf16(
                    false, a.v, false, bfr.v, (short)0, acc[nsub], false, false);
            }
        }
        __syncthreads();
    }

    // epilogue: D layout (n = lane&15, m = (lane>>4)*8 + vgpr) -> Gout[m][n]
    const int nbase = n0 + (lane & 15);
    const int mbase = m0 + wave * 16 + (lane >> 4) * 8;
#pragma unroll
    for (int nsub = 0; nsub < 4; ++nsub) {
        int nn = nbase + nsub * 16;
#pragma unroll
        for (int r = 0; r < 8; ++r)
            Gout[(size_t)(mbase + r) * NS + nn] = acc[nsub][r];
    }
}

// -------------------------------------------------------- pointwise LSTM L1
__global__ void stff_pw1(const float* __restrict__ G, const float* __restrict__ bx,
                         const float* __restrict__ Wp, float* __restrict__ cst,
                         u16* __restrict__ I1, u16* __restrict__ I2) {
    int tid = blockIdx.x * 256 + threadIdx.x;
    if (tid >= HID1 * NPOS) return;
    int c = tid / NPOS;
    int n = tid - c * NPOS;
    int b = n / (H_IN * W_IN);
    int pos = n - b * (H_IN * W_IN);
    int y = pos / W_IN, xx = pos - y * W_IN;

    float gi = G[c * NS + n]              + bx[c];
    float gf = G[(HID1 + c) * NS + n]     + bx[HID1 + c];
    float gg = G[(2 * HID1 + c) * NS + n] + bx[2 * HID1 + c];
    float go = G[(3 * HID1 + c) * NS + n] + bx[3 * HID1 + c];

    int cidx = (b * HID1 + c) * (H_IN * W_IN) + pos;
    float cp = cst[cidx];
    int pp = c * (H_IN * W_IN) + pos;
    float i = sigm(gi + cp * Wp[pp]);
    float f = sigm(gf + cp * Wp[HID1 * H_IN * W_IN + pp]);
    float cc = f * cp + i * tanhf(gg);
    float o = sigm(go + cc * Wp[2 * HID1 * H_IN * W_IN + pp]);
    float h = o * tanhf(cc);
    cst[cidx] = cc;

    u16 hb = f2bf(h);
    I1[((b * CIN + 1 + c) * HP + y + 1) * WPAD + xx + 1] = hb;  // next-step L1 input
    I2[((b * CIN + c) * HP + y + 1) * WPAD + xx + 1]     = hb;  // this-step L2 input
}

// -------------------------------------------------------- pointwise LSTM L2
__global__ void stff_pw2(const float* __restrict__ G, const float* __restrict__ bx,
                         const float* __restrict__ Wp, float* __restrict__ cst,
                         u16* __restrict__ I2, float* __restrict__ h2f) {
    int tid = blockIdx.x * 256 + threadIdx.x;
    if (tid >= HID2 * NPOS) return;
    int c = tid / NPOS;
    int n = tid - c * NPOS;
    int b = n / (H_IN * W_IN);
    int pos = n - b * (H_IN * W_IN);
    int y = pos / W_IN, xx = pos - y * W_IN;

    float gi = G[c * NS + n]              + bx[c];
    float gf = G[(HID2 + c) * NS + n]     + bx[HID2 + c];
    float gg = G[(2 * HID2 + c) * NS + n] + bx[2 * HID2 + c];
    float go = G[(3 * HID2 + c) * NS + n] + bx[3 * HID2 + c];

    int cidx = (b * HID2 + c) * (H_IN * W_IN) + pos;
    float cp = cst[cidx];
    int pp = c * (H_IN * W_IN) + pos;
    float i = sigm(gi + cp * Wp[pp]);
    float f = sigm(gf + cp * Wp[HID2 * H_IN * W_IN + pp]);
    float cc = f * cp + i * tanhf(gg);
    float o = sigm(go + cc * Wp[2 * HID2 * H_IN * W_IN + pp]);
    float h = o * tanhf(cc);
    cst[cidx] = cc;

    I2[((b * CIN + 128 + c) * HP + y + 1) * WPAD + xx + 1] = f2bf(h);
    h2f[cidx] = h;
}

// ------------------------------------------------------------- decoder convT1
__global__ void stff_convt1(const float* __restrict__ h2, const float* __restrict__ Wt1,
                            const float* __restrict__ bt1, float* __restrict__ out1) {
    int tid = blockIdx.x * 256 + threadIdx.x;
    if (tid >= BATCH * 16 * H1OUT * W1OUT) return;
    int xx = tid % W1OUT;
    int y  = (tid / W1OUT) % H1OUT;
    int o  = (tid / (W1OUT * H1OUT)) % 16;
    int b  = tid / (W1OUT * H1OUT * 16);
    float acc = bt1[o];
    for (int c = 0; c < HID2; ++c) {
        const float* hb = h2 + (b * HID2 + c) * (H_IN * W_IN);
#pragma unroll
        for (int ky = 0; ky < 3; ++ky) {
            int ty = y - ky;
            if (ty < 0 || (ty & 1)) continue;
            int iy = ty >> 1;
            if (iy >= H_IN) continue;
#pragma unroll
            for (int kx = 0; kx < 3; ++kx) {
                int tx = xx - kx;
                if (tx < 0 || (tx & 1)) continue;
                int ix = tx >> 1;
                if (ix >= W_IN) continue;
                acc += hb[iy * W_IN + ix] * Wt1[((c * 16 + o) * 3 + ky) * 3 + kx];
            }
        }
    }
    out1[tid] = fmaxf(acc, 0.0f);
}

// ------------------------------------------------------------- decoder convT2
__global__ void stff_convt2(const float* __restrict__ out1, const float* __restrict__ Wt2,
                            const float* __restrict__ bt2, float* __restrict__ out2) {
    int tid = blockIdx.x * 256 + threadIdx.x;
    if (tid >= BATCH * H2OUT * W2OUT) return;
    int xx = tid % W2OUT;
    int y  = (tid / W2OUT) % H2OUT;
    int b  = tid / (W2OUT * H2OUT);
    float acc = bt2[0];
    for (int c = 0; c < 16; ++c) {
        const float* ob = out1 + (b * 16 + c) * (H1OUT * W1OUT);
#pragma unroll
        for (int ky = 0; ky < 5; ++ky) {
            int ty = y - ky;
            if (ty < 0 || (ty & 1)) continue;
            int iy = ty >> 1;
            if (iy >= H1OUT) continue;
#pragma unroll
            for (int kx = 0; kx < 5; ++kx) {
                int tx = xx - kx;
                if (tx < 0 || (tx & 1)) continue;
                int ix = tx >> 1;
                if (ix >= W1OUT) continue;
                acc += ob[iy * W1OUT + ix] * Wt2[c * 25 + ky * 5 + kx];
            }
        }
    }
    out2[tid] = acc;
}

// ----------------------------------------------------------------- avg pool
__global__ void stff_pool(const float* __restrict__ out2, float* __restrict__ out) {
    int tid = blockIdx.x * 256 + threadIdx.x;
    if (tid >= BATCH * HPOOL * WPOOL) return;
    int p = tid % WPOOL;
    int o = (tid / WPOOL) % HPOOL;
    int b = tid / (WPOOL * HPOOL);
    int sh = (o * H2OUT) / HPOOL;
    int eh = ((o + 1) * H2OUT + HPOOL - 1) / HPOOL;
    int sw = (p * W2OUT) / WPOOL;
    int ew = ((p + 1) * W2OUT + WPOOL - 1) / WPOOL;
    float wh = 1.0f / (float)(eh - sh);
    float ww = 1.0f / (float)(ew - sw);
    float acc = 0.0f;
    for (int h = sh; h < eh; ++h)
        for (int w = sw; w < ew; ++w)
            acc += out2[(b * H2OUT + h) * W2OUT + w];
    out[tid] = acc * wh * ww;
}

// ================================================================= launcher
extern "C" void kernel_launch(void* const* d_in, const int* in_sizes, int n_in,
                              void* d_out, int out_size, void* d_ws, size_t ws_size,
                              hipStream_t stream) {
    const float* x   = (const float*)d_in[0];
    const float* Wx1 = (const float*)d_in[1];
    const float* bx1 = (const float*)d_in[2];
    const float* Wh1 = (const float*)d_in[3];
    const float* Wp1 = (const float*)d_in[4];
    const float* Wx2 = (const float*)d_in[5];
    const float* bx2 = (const float*)d_in[6];
    const float* Wh2 = (const float*)d_in[7];
    const float* Wp2 = (const float*)d_in[8];
    const float* Wt1 = (const float*)d_in[9];
    const float* bt1 = (const float*)d_in[10];
    const float* Wt2 = (const float*)d_in[11];
    const float* bt2 = (const float*)d_in[12];
    float* out = (float*)d_out;

    char* ws = (char*)d_ws;
    size_t off = 0;
    auto alloc = [&](size_t bytes) -> void* {
        void* p = ws + off;
        off += (bytes + 255) & ~(size_t)255;
        return p;
    };
    const int IBUF = BATCH * CIN * HP * WPAD;   // 819200 bf16 elems
    u16*   I1    = (u16*)  alloc((size_t)IBUF * 2);
    u16*   I2    = (u16*)  alloc((size_t)IBUF * 2);
    u16*   W1cat = (u16*)  alloc((size_t)M1 * KDIM * 2);
    u16*   W2cat = (u16*)  alloc((size_t)M2 * KDIM * 2);
    u16*   Bcol  = (u16*)  alloc((size_t)NS * KDIM * 2);   // shared by both layers
    float* G1    = (float*)alloc((size_t)M1 * NS * 4);
    float* G2    = (float*)alloc((size_t)M2 * NS * 4);
    float* c1    = (float*)alloc((size_t)BATCH * HID1 * H_IN * W_IN * 4);
    float* c2    = (float*)alloc((size_t)BATCH * HID2 * H_IN * W_IN * 4);
    float* h2f   = (float*)alloc((size_t)BATCH * HID2 * H_IN * W_IN * 4);
    float* o1    = (float*)alloc((size_t)BATCH * 16 * H1OUT * W1OUT * 4);
    float* o2    = (float*)alloc((size_t)BATCH * H2OUT * W2OUT * 4);

    auto gridOf = [](int n) { return dim3((n + 255) / 256); };

    // zero state every call (harness does not re-poison between replays)
    stff_zero_u16<<<gridOf(IBUF), 256, 0, stream>>>(I1, IBUF);
    stff_zero_u16<<<gridOf(IBUF), 256, 0, stream>>>(I2, IBUF);
    stff_zero_f32<<<gridOf(BATCH * HID1 * H_IN * W_IN), 256, 0, stream>>>(c1, BATCH * HID1 * H_IN * W_IN);
    stff_zero_f32<<<gridOf(BATCH * HID2 * H_IN * W_IN), 256, 0, stream>>>(c2, BATCH * HID2 * H_IN * W_IN);
    stff_zero_f32<<<gridOf(BATCH * HID2 * H_IN * W_IN), 256, 0, stream>>>(h2f, BATCH * HID2 * H_IN * W_IN);

    // pack weights to bf16 im2col layout
    stff_prep_w1<<<gridOf(M1 * KDIM), 256, 0, stream>>>(Wx1, Wh1, W1cat);
    stff_prep_w2<<<gridOf(M2 * KDIM), 256, 0, stream>>>(Wx2, Wh2, W2cat);

    const dim3 gemmBlk(128);
    const dim3 g1(M1 / 64, NS / 64);
    const dim3 g2(M2 / 64, NS / 64);
    const dim3 colGrid((NS * CIN + 255) / 256);

    for (int t = 0; t < TSTEPS; ++t) {
        stff_stage_x<<<gridOf(NPOS), 256, 0, stream>>>(x, I1, t);
        stff_im2col<<<colGrid, 256, 0, stream>>>(I1, Bcol);
        stff_gate_gemm<<<g1, gemmBlk, 0, stream>>>(W1cat, Bcol, G1);
        stff_pw1<<<gridOf(HID1 * NPOS), 256, 0, stream>>>(G1, bx1, Wp1, c1, I1, I2);
        stff_im2col<<<colGrid, 256, 0, stream>>>(I2, Bcol);
        stff_gate_gemm<<<g2, gemmBlk, 0, stream>>>(W2cat, Bcol, G2);
        stff_pw2<<<gridOf(HID2 * NPOS), 256, 0, stream>>>(G2, bx2, Wp2, c2, I2, h2f);
    }

    stff_convt1<<<gridOf(BATCH * 16 * H1OUT * W1OUT), 256, 0, stream>>>(h2f, Wt1, bt1, o1);
    stff_convt2<<<gridOf(BATCH * H2OUT * W2OUT), 256, 0, stream>>>(o1, Wt2, bt2, o2);
    stff_pool<<<gridOf(BATCH * HPOOL * WPOOL), 256, 0, stream>>>(o2, out);
}
